// LiquidTransformerLayer_52922587022083
// MI455X (gfx1250) — matrix operand order repaired and checked by reading, not compiled, not run on hardware
//
#include <hip/hip_runtime.h>
#include <hip/hip_bf16.h>

// ---------------------------------------------------------------------------
// Types for gfx1250 WMMA (wave32): v_wmma_f32_16x16x32_bf16
// ---------------------------------------------------------------------------
typedef __attribute__((ext_vector_type(16))) __bf16 v16bf;
typedef __attribute__((ext_vector_type(8)))  float  v8f;

union FragU { uint4 u[2]; v16bf v; };

static_assert(sizeof(v16bf) == 32, "v16bf must be 32 bytes");

constexpr int EPI_F32       = 0;  // C = A@B (+bias)          -> f32
constexpr int EPI_BF16      = 1;  // C = A@B                  -> bf16
constexpr int EPI_TANH_BF16 = 2;  // C = tanh(A@B)            -> bf16
constexpr int EPI_RESID_F32 = 3;  // C = A@B + bias + resid   -> f32

// ---------------------------------------------------------------------------
// CDNA5 async global->LDS copy (ASYNCcnt-tracked). Low 32 bits of a generic
// pointer to a __shared__ object are the LDS byte address (aperture rule).
// ---------------------------------------------------------------------------
__device__ __forceinline__ void async_copy_b128(const void* gptr, void* lptr) {
  const uint32_t lds = (uint32_t)(uintptr_t)lptr;
  asm volatile("global_load_async_to_lds_b128 %0, %1, off"
               :: "v"(lds), "v"(gptr)
               : "memory");
}
__device__ __forceinline__ void wait_async0() {
  asm volatile("s_wait_asynccnt 0x0" ::: "memory");
}

// ---------------------------------------------------------------------------
// f32 -> bf16 conversion
// ---------------------------------------------------------------------------
__global__ void f32_to_bf16_kernel(const float* __restrict__ in,
                                   __bf16* __restrict__ out, int n) {
  int i = blockIdx.x * blockDim.x + threadIdx.x;
  if (i < n) out[i] = (__bf16)in[i];
}

// ---------------------------------------------------------------------------
// LayerNorm over rows of 1024; emits bf16 normalized activations.
// ---------------------------------------------------------------------------
__global__ __launch_bounds__(256)
void layernorm_kernel(const float* __restrict__ x, const float* __restrict__ g,
                      const float* __restrict__ bb, __bf16* __restrict__ out) {
  __shared__ float red[8];
  const int row = blockIdx.x;
  const int tid = threadIdx.x;
  const float* xr = x + (size_t)row * 1024;
  float4 v = reinterpret_cast<const float4*>(xr)[tid];
  float s = v.x + v.y + v.z + v.w;
#pragma unroll
  for (int off = 16; off > 0; off >>= 1) s += __shfl_xor(s, off, 32);
  if ((tid & 31) == 0) red[tid >> 5] = s;
  __syncthreads();
  float mu = 0.f;
#pragma unroll
  for (int i = 0; i < 8; ++i) mu += red[i];
  mu *= (1.f / 1024.f);
  __syncthreads();
  const float dx = v.x - mu, dy = v.y - mu, dz = v.z - mu, dw = v.w - mu;
  float ss = dx * dx + dy * dy + dz * dz + dw * dw;
#pragma unroll
  for (int off = 16; off > 0; off >>= 1) ss += __shfl_xor(ss, off, 32);
  if ((tid & 31) == 0) red[tid >> 5] = ss;
  __syncthreads();
  float var = 0.f;
#pragma unroll
  for (int i = 0; i < 8; ++i) var += red[i];
  var *= (1.f / 1024.f);
  const float rstd = rsqrtf(var + 1e-5f);
  const int c = tid * 4;
  __bf16* op = out + (size_t)row * 1024 + c;
  op[0] = (__bf16)(dx * rstd * g[c + 0] + bb[c + 0]);
  op[1] = (__bf16)(dy * rstd * g[c + 1] + bb[c + 1]);
  op[2] = (__bf16)(dz * rstd * g[c + 2] + bb[c + 2]);
  op[3] = (__bf16)(dw * rstd * g[c + 3] + bb[c + 3]);
}

// ---------------------------------------------------------------------------
// Generic WMMA bf16 GEMM: C[M,N] = A[M,K] @ B[K,N] with fused epilogues.
// Block tile 128x64, 8 waves (4x2), wave tile 32x32 (2x2 wmma tiles), K-step 32.
// A tile staged by async global->LDS copies; B tile transposed manually.
// LDS double-buffered: tile k+1 streams in while tile k feeds the WMMAs.
// ---------------------------------------------------------------------------
template <int MODE>
__global__ __launch_bounds__(256)
void gemm_bf16_kernel(const __bf16* __restrict__ A, const __bf16* __restrict__ Bm,
                      void* __restrict__ Cout, const float* __restrict__ bias,
                      const float* __restrict__ resid, int M, int N, int K) {
  __shared__ __attribute__((aligned(16))) __bf16 As[2][128 * 40];
  __shared__ __attribute__((aligned(16))) __bf16 Bs[2][64 * 40];

  const int tid  = threadIdx.x;
  const int lane = tid & 31;
  const int wave = tid >> 5;
  const int wm   = wave >> 1;   // 0..3
  const int wn   = wave & 1;    // 0..1
  const int bm   = blockIdx.y * 128;
  const int bn   = blockIdx.x * 64;
  const int lh   = lane & 15;
  const int hi   = lane >> 4;   // 0 or 1

  v8f acc[2][2] = {};

  auto load_tile = [&](int k0, int buf) {
    // A tile: 128x32 bf16 = 512 x 16B chunks, 2 async copies per thread
#pragma unroll
    for (int i = 0; i < 2; ++i) {
      const int c  = tid + i * 256;
      const int r  = c >> 2;
      const int kc = (c & 3) * 8;
      async_copy_b128(A + (size_t)(bm + r) * K + k0 + kc, &As[buf][r * 40 + kc]);
    }
    // B tile: 32x64, transposed into LDS [n][k]
    const int kr = tid >> 3;        // 0..31
    const int nc = (tid & 7) * 8;   // 0..56
    const uint4 v = *reinterpret_cast<const uint4*>(Bm + (size_t)(k0 + kr) * N + bn + nc);
    const __bf16* e = reinterpret_cast<const __bf16*>(&v);
#pragma unroll
    for (int j = 0; j < 8; ++j) Bs[buf][(nc + j) * 40 + kr] = e[j];
    // gfx1250 speculative prefetch of the k+2 B slab (global_prefetch_b8)
    if (k0 + 32 < K)
      __builtin_prefetch(Bm + (size_t)(k0 + 32 + kr) * N + bn + nc, 0, 1);
  };

  const int NK = K >> 5;
  load_tile(0, 0);

  for (int kt = 0; kt < NK; ++kt) {
    const int cur = kt & 1;
    wait_async0();        // our async writes for tile kt have landed in LDS
    __syncthreads();      // everyone's writes visible; prior reads retired
    if (kt + 1 < NK) load_tile((kt + 1) << 5, cur ^ 1);

    // --- fragments from buffer `cur` ---
    v16bf afrag[2], bfrag[2];
#pragma unroll
    for (int mt = 0; mt < 2; ++mt) {
      const int row = wm * 32 + mt * 16 + lh;
      const int kh  = hi * 8;
      FragU f;
      f.u[0] = *reinterpret_cast<const uint4*>(&As[cur][row * 40 + kh]);
      f.u[1] = *reinterpret_cast<const uint4*>(&As[cur][row * 40 + kh + 16]);
      afrag[mt] = f.v;
    }
#pragma unroll
    for (int nt = 0; nt < 2; ++nt) {
      const int col = wn * 32 + nt * 16 + lh;
      const int kk  = hi * 8;
      FragU f;
      f.u[0] = *reinterpret_cast<const uint4*>(&Bs[cur][col * 40 + kk]);
      f.u[1] = *reinterpret_cast<const uint4*>(&Bs[cur][col * 40 + kk + 16]);
      bfrag[nt] = f.v;
    }
#pragma unroll
    for (int mt = 0; mt < 2; ++mt)
#pragma unroll
      for (int nt = 0; nt < 2; ++nt)
        acc[mt][nt] = __builtin_amdgcn_wmma_f32_16x16x32_bf16(
            false, afrag[mt], false, bfrag[nt], (short)0, acc[mt][nt], false, false);
  }

  // --- epilogue (C layout: VGPR v -> row v + 8*hi, col = lane&15) ---
#pragma unroll
  for (int mt = 0; mt < 2; ++mt)
#pragma unroll
    for (int nt = 0; nt < 2; ++nt)
#pragma unroll
      for (int v = 0; v < 8; ++v) {
        const int row = bm + wm * 32 + mt * 16 + hi * 8 + v;
        const int col = bn + wn * 32 + nt * 16 + lh;
        const size_t idx = (size_t)row * N + col;
        float val = acc[mt][nt][v];
        if (MODE == EPI_F32) {
          if (bias) val += bias[col];
          reinterpret_cast<float*>(Cout)[idx] = val;
        } else if (MODE == EPI_BF16) {
          reinterpret_cast<__bf16*>(Cout)[idx] = (__bf16)val;
        } else if (MODE == EPI_TANH_BF16) {
          reinterpret_cast<__bf16*>(Cout)[idx] = (__bf16)tanhf(val);
        } else {  // EPI_RESID_F32
          val += bias[col] + resid[idx];
          reinterpret_cast<float*>(Cout)[idx] = val;
        }
      }
}

// ---------------------------------------------------------------------------
// Flash attention (WMMA). Grid: x = L/64 query blocks, y = B*NH.
// Block: 128 threads = 4 waves; each wave owns 16 queries, dh = 64.
// Q and K tiles staged via async global->LDS copies; V transposed manually.
// ---------------------------------------------------------------------------
__global__ __launch_bounds__(128)
void flash_attn_kernel(const __bf16* __restrict__ Q, const __bf16* __restrict__ Kg,
                       const __bf16* __restrict__ Vg, __bf16* __restrict__ O) {
  constexpr int L = 2048, H = 1024, DH = 64, KS = 72;
  __shared__ __attribute__((aligned(16))) __bf16 Kl[64 * KS];
  __shared__ __attribute__((aligned(16))) __bf16 Vt[64 * KS];
  __shared__ __attribute__((aligned(16))) __bf16 Pl[4 * 16 * KS];

  const int tid  = threadIdx.x;
  const int lane = tid & 31;
  const int wave = tid >> 5;
  const int lh   = lane & 15;
  const int hi   = lane >> 4;
  const int qb   = blockIdx.x * 64;
  const int bh   = blockIdx.y;
  const int b    = bh >> 4;
  const int h    = bh & 15;
  const size_t baseQ = ((size_t)b * L + qb) * H + h * DH;

  // --- stage Q tile through Kl (async), build per-wave Q fragments ---
#pragma unroll
  for (int i = 0; i < 4; ++i) {
    const int c  = tid + i * 128;
    const int r  = c >> 3;
    const int dc = (c & 7) * 8;
    async_copy_b128(Q + baseQ + (size_t)r * H + dc, &Kl[r * KS + dc]);
  }
  wait_async0();
  __syncthreads();
  v16bf qf[2];
  {
    const int row = wave * 16 + lh;
    const int kh  = hi * 8;
#pragma unroll
    for (int ks = 0; ks < 2; ++ks) {
      FragU f;
      f.u[0] = *reinterpret_cast<const uint4*>(&Kl[row * KS + ks * 32 + kh]);
      f.u[1] = *reinterpret_cast<const uint4*>(&Kl[row * KS + ks * 32 + kh + 16]);
      qf[ks] = f.v;
    }
  }
  __syncthreads();

  float m[8], l[8];
  v8f o[4] = {};
#pragma unroll
  for (int v = 0; v < 8; ++v) { m[v] = -1e30f; l[v] = 0.f; }

  for (int kb = 0; kb < L; kb += 64) {
    const size_t baseK = ((size_t)b * L + kb) * H + h * DH;
    // --- K tile: async straight copy; V tile: manual transpose ---
#pragma unroll
    for (int i = 0; i < 4; ++i) {
      const int c  = tid + i * 128;
      const int r  = c >> 3;
      const int dc = (c & 7) * 8;
      async_copy_b128(Kg + baseK + (size_t)r * H + dc, &Kl[r * KS + dc]);
      const uint4 vv = *reinterpret_cast<const uint4*>(Vg + baseK + (size_t)r * H + dc);
      const __bf16* e = reinterpret_cast<const __bf16*>(&vv);
#pragma unroll
      for (int j = 0; j < 8; ++j) Vt[(dc + j) * KS + r] = e[j];
    }
    wait_async0();
    __syncthreads();

    // --- S = Q @ K^T : 4 kv tiles of 16, K-dim = dh = 64 = 2 wmma steps ---
    v8f s[4];
#pragma unroll
    for (int kt = 0; kt < 4; ++kt) {
      v8f a = {};
#pragma unroll
      for (int ks = 0; ks < 2; ++ks) {
        const int col = kt * 16 + lh;          // kv index (B-matrix N)
        const int kk  = ks * 32 + hi * 8;      // d offset (B-matrix K)
        FragU f;
        f.u[0] = *reinterpret_cast<const uint4*>(&Kl[col * KS + kk]);
        f.u[1] = *reinterpret_cast<const uint4*>(&Kl[col * KS + kk + 16]);
        a = __builtin_amdgcn_wmma_f32_16x16x32_bf16(false, qf[ks], false, f.v,
                                                    (short)0, a, false, false);
      }
      s[kt] = a;
    }

    // --- online softmax (row of S spans 16 lanes in one VGPR slot) ---
    float pn[4][8];
#pragma unroll
    for (int v = 0; v < 8; ++v) {
      float mx = -1e30f;
#pragma unroll
      for (int kt = 0; kt < 4; ++kt) mx = fmaxf(mx, s[kt][v] * 0.125f);
#pragma unroll
      for (int off = 1; off < 16; off <<= 1) mx = fmaxf(mx, __shfl_xor(mx, off, 32));
      const float mnew = fmaxf(m[v], mx);
      const float corr = __expf(m[v] - mnew);
      float rs = 0.f;
#pragma unroll
      for (int kt = 0; kt < 4; ++kt) {
        const float p = __expf(s[kt][v] * 0.125f - mnew);
        pn[kt][v] = p;
        rs += p;
      }
#pragma unroll
      for (int off = 1; off < 16; off <<= 1) rs += __shfl_xor(rs, off, 32);
      l[v] = l[v] * corr + rs;
      m[v] = mnew;
#pragma unroll
      for (int dt = 0; dt < 4; ++dt) o[dt][v] *= corr;
    }

    // --- P: C-layout regs -> per-wave LDS -> A-layout fragments ---
    __bf16* P = &Pl[wave * 16 * KS];
#pragma unroll
    for (int v = 0; v < 8; ++v) {
      const int row = hi * 8 + v;
#pragma unroll
      for (int kt = 0; kt < 4; ++kt)
        P[row * KS + kt * 16 + lh] = (__bf16)pn[kt][v];
    }

    // --- O += P @ V ---
#pragma unroll
    for (int ks = 0; ks < 2; ++ks) {
      FragU pf;
      pf.u[0] = *reinterpret_cast<const uint4*>(&P[lh * KS + ks * 32 + hi * 8]);
      pf.u[1] = *reinterpret_cast<const uint4*>(&P[lh * KS + ks * 32 + hi * 8 + 16]);
#pragma unroll
      for (int dt = 0; dt < 4; ++dt) {
        const int col = dt * 16 + lh;          // d index (B-matrix N)
        const int kk  = ks * 32 + hi * 8;      // kv offset (B-matrix K)
        FragU vf;
        vf.u[0] = *reinterpret_cast<const uint4*>(&Vt[col * KS + kk]);
        vf.u[1] = *reinterpret_cast<const uint4*>(&Vt[col * KS + kk + 16]);
        o[dt] = __builtin_amdgcn_wmma_f32_16x16x32_bf16(false, pf.v, false, vf.v,
                                                        (short)0, o[dt], false, false);
      }
    }
    __syncthreads();
  }

  // --- normalize and write bf16 output [B*L, H] ---
#pragma unroll
  for (int v = 0; v < 8; ++v) {
    const float inv = 1.f / l[v];
    const int q = qb + wave * 16 + hi * 8 + v;
#pragma unroll
    for (int dt = 0; dt < 4; ++dt) {
      const int col = h * DH + dt * 16 + lh;
      O[((size_t)b * L + q) * H + col] = (__bf16)(o[dt][v] * inv);
    }
  }
}

// ---------------------------------------------------------------------------
// gate = xn2 @ Wg + bg (f32 [4096,4096]); h = a * sigmoid(b) -> bf16 [4096,2048]
// ---------------------------------------------------------------------------
__global__ void gate_act_kernel(const float* __restrict__ gate,
                                __bf16* __restrict__ h, int n) {
  const int i = blockIdx.x * blockDim.x + threadIdx.x;
  if (i >= n) return;
  const int r = i >> 11;
  const int c = i & 2047;
  const float a  = gate[(size_t)r * 4096 + c];
  const float bb = gate[(size_t)r * 4096 + 2048 + c];
  h[i] = (__bf16)(a * (1.f / (1.f + __expf(-bb))));
}

// ---------------------------------------------------------------------------
// Host orchestration
// ---------------------------------------------------------------------------
extern "C" void kernel_launch(void* const* d_in, const int* in_sizes, int n_in,
                              void* d_out, int out_size, void* d_ws, size_t ws_size,
                              hipStream_t stream) {
  (void)in_sizes; (void)n_in; (void)out_size; (void)ws_size;
  const float* x   = (const float*)d_in[0];
  const float* Wq  = (const float*)d_in[1];
  const float* Wk  = (const float*)d_in[2];
  const float* Wv  = (const float*)d_in[3];
  const float* Wo  = (const float*)d_in[4];
  const float* bo  = (const float*)d_in[5];
  const float* Wrq = (const float*)d_in[6];
  const float* Wrk = (const float*)d_in[7];
  const float* Wg  = (const float*)d_in[8];
  const float* bg  = (const float*)d_in[9];
  const float* Wf  = (const float*)d_in[10];
  const float* bf  = (const float*)d_in[11];
  const float* g1  = (const float*)d_in[12];
  const float* b1  = (const float*)d_in[13];
  const float* g2  = (const float*)d_in[14];
  const float* b2  = (const float*)d_in[15];

  const int M = 4096;   // B * L
  const int H = 1024;

  char* w = (char*)d_ws;
  size_t off = 0;
  auto carve = [&](size_t bytes) -> void* {
    void* p = w + off;
    off += (bytes + 255) & ~(size_t)255;
    return p;
  };

  __bf16* Wq_b  = (__bf16*)carve((size_t)H * H * 2);
  __bf16* Wk_b  = (__bf16*)carve((size_t)H * H * 2);
  __bf16* Wv_b  = (__bf16*)carve((size_t)H * H * 2);
  __bf16* Wo_b  = (__bf16*)carve((size_t)H * H * 2);
  __bf16* Wrq_b = (__bf16*)carve((size_t)H * H * 2);
  __bf16* Wrk_b = (__bf16*)carve((size_t)H * H * 2);
  __bf16* Wg_b  = (__bf16*)carve((size_t)H * 4 * H * 2);
  __bf16* Wf_b  = (__bf16*)carve((size_t)2 * H * H * 2);
  __bf16* xn_b  = (__bf16*)carve((size_t)M * H * 2);
  __bf16* t1_b  = (__bf16*)carve((size_t)M * H * 2);
  __bf16* Q_b   = (__bf16*)carve((size_t)M * H * 2);
  __bf16* K_b   = (__bf16*)carve((size_t)M * H * 2);
  __bf16* V_b   = (__bf16*)carve((size_t)M * H * 2);
  __bf16* at_b  = (__bf16*)carve((size_t)M * H * 2);
  float*  x1    = (float*) carve((size_t)M * H * 4);
  __bf16* xn2_b = (__bf16*)carve((size_t)M * H * 2);
  float*  gate  = (float*) carve((size_t)M * 4 * H * 4);
  __bf16* h_b   = (__bf16*)carve((size_t)M * 2 * H * 2);

  auto conv = [&](const float* src, __bf16* dst, int n) {
    f32_to_bf16_kernel<<<(n + 255) / 256, 256, 0, stream>>>(src, dst, n);
  };
  conv(Wq, Wq_b, H * H);
  conv(Wk, Wk_b, H * H);
  conv(Wv, Wv_b, H * H);
  conv(Wo, Wo_b, H * H);
  conv(Wrq, Wrq_b, H * H);
  conv(Wrk, Wrk_b, H * H);
  conv(Wg, Wg_b, H * 4 * H);
  conv(Wf, Wf_b, 2 * H * H);

  const dim3 blk(256);
  auto grid = [](int N_, int M_) { return dim3(N_ / 64, M_ / 128); };

  // LN1
  layernorm_kernel<<<M, 256, 0, stream>>>(x, g1, b1, xn_b);
  // Q = tanh((xn @ Wq) @ Wres_q)
  gemm_bf16_kernel<EPI_BF16><<<grid(H, M), blk, 0, stream>>>(xn_b, Wq_b, t1_b, nullptr, nullptr, M, H, H);
  gemm_bf16_kernel<EPI_TANH_BF16><<<grid(H, M), blk, 0, stream>>>(t1_b, Wrq_b, Q_b, nullptr, nullptr, M, H, H);
  // K = tanh((xn @ Wk) @ Wres_k)
  gemm_bf16_kernel<EPI_BF16><<<grid(H, M), blk, 0, stream>>>(xn_b, Wk_b, t1_b, nullptr, nullptr, M, H, H);
  gemm_bf16_kernel<EPI_TANH_BF16><<<grid(H, M), blk, 0, stream>>>(t1_b, Wrk_b, K_b, nullptr, nullptr, M, H, H);
  // V = xn @ Wv
  gemm_bf16_kernel<EPI_BF16><<<grid(H, M), blk, 0, stream>>>(xn_b, Wv_b, V_b, nullptr, nullptr, M, H, H);
  // attention (32 query blocks x 32 (batch,head) pairs)
  flash_attn_kernel<<<dim3(32, 32), 128, 0, stream>>>(Q_b, K_b, V_b, at_b);
  // x1 = x + attn @ Wo + bo
  gemm_bf16_kernel<EPI_RESID_F32><<<grid(H, M), blk, 0, stream>>>(at_b, Wo_b, x1, bo, x, M, H, H);
  // LN2
  layernorm_kernel<<<M, 256, 0, stream>>>(x1, g2, b2, xn2_b);
  // gate = xn2 @ Wg + bg
  gemm_bf16_kernel<EPI_F32><<<grid(4 * H, M), blk, 0, stream>>>(xn2_b, Wg_b, gate, bg, nullptr, M, 4 * H, H);
  // h = a * sigmoid(b)
  gate_act_kernel<<<(M * 2 * H + 255) / 256, 256, 0, stream>>>(gate, h_b, M * 2 * H);
  // out = x1 + h @ Wf + bf
  gemm_bf16_kernel<EPI_RESID_F32><<<grid(H, M), blk, 0, stream>>>(h_b, Wf_b, d_out, bf, x1, M, H, 2 * H);
}
